// QKVOrthoPosNegPromptG_89412629168761
// MI455X (gfx1250) — matrix operand order repaired and checked by reading, not compiled, not run on hardware
//
#include <hip/hip_runtime.h>
#include <hip/hip_bf16.h>

// ---------------------------------------------------------------------------
// Problem constants (from reference)
// ---------------------------------------------------------------------------
#define BB   16
#define SS   4096
#define DD   768
#define HH   8
#define EE   768
#define LQ   48
#define HD   96
// 1/sqrt(96)
#define QK_SCALE 0.10206207261596575f

typedef unsigned short u16;
typedef unsigned int   u32;
typedef unsigned long long u64;

// WMMA vector types (CDNA5 gfx1250, wave32)
typedef __attribute__((ext_vector_type(16))) __bf16 v16bf;
typedef __attribute__((ext_vector_type(8)))  float  v8f;
// TDM descriptor groups (clang-23 6-arg builtin form)
typedef __attribute__((ext_vector_type(4)))  u32 v4u;
typedef __attribute__((ext_vector_type(8)))  int v8i;
typedef __attribute__((ext_vector_type(4)))  int v4i;

union FragBf16 { v16bf v; uint4 q[2]; };

#if __has_builtin(__builtin_amdgcn_tensor_load_to_lds) && \
    __has_builtin(__builtin_amdgcn_s_wait_tensorcnt)
#define USE_TDM 1
#else
#define USE_TDM 0
#endif

#define AS3 __attribute__((address_space(3)))

// ---------------------------------------------------------------------------
// small helpers
// ---------------------------------------------------------------------------
__device__ __forceinline__ u16 f2bf(float f) {
    union { float f; u32 i; } x; x.f = f;
    u32 r = x.i + 0x7FFFu + ((x.i >> 16) & 1u);   // RNE
    return (u16)(r >> 16);
}
__device__ __forceinline__ float bf2f(u16 u) {
    union { u32 i; float f; } x; x.i = ((u32)u) << 16;
    return x.f;
}
__device__ __forceinline__ float bits2f(u32 b) {
    union { u32 i; float f; } x; x.i = b; return x.f;
}

#if USE_TDM
// LDS byte address of a __shared__ object (addrspacecast generic -> AS3)
__device__ __forceinline__ u32 lds_byte_addr(const void* p) {
    return (u32)(u64)(AS3 const void*)p;
}
// Issue one TDM 2D tile load: tile_h rows x tile_w bf16 elements, row stride
// stride_elems (in elements), packed contiguously into LDS at ldsAddr.
// Descriptor layout per CDNA5 ISA section 8 (D# groups 0/1; 2D tensor).
__device__ __forceinline__ void tdm_load_2d_bf16(u32 ldsAddr, const void* gptr,
                                                 int stride_elems,
                                                 int tile_w, int tile_h) {
    u64 ga = (u64)gptr;
    v4u g0;
    g0[0] = 1u;                                      // count=1, user mode
    g0[1] = ldsAddr;                                 // lds_addr (bytes)
    g0[2] = (u32)(ga & 0xFFFFFFFFu);                 // global_addr[31:0]
    g0[3] = (u32)((ga >> 32) & 0x1FFFFFFu)           // global_addr[56:32]
          | (2u << 30);                              // type=2 ("image")
    v8i g1;
    g1[0] = (int)(1u << 16);                         // wg_mask=0, data_size=1 (2B)
    g1[1] = (int)(((u32)tile_w & 0xFFFFu) << 16);    // tensor_dim0[15:0]
    g1[2] = (int)((((u32)tile_w >> 16) & 0xFFFFu)    // tensor_dim0[31:16]
          | (((u32)tile_h & 0xFFFFu) << 16));        // tensor_dim1[15:0]
    g1[3] = (int)((((u32)tile_h >> 16) & 0xFFFFu)    // tensor_dim1[31:16]
          | (((u32)tile_w & 0xFFFFu) << 16));        // tile_dim0
    g1[4] = (int)((u32)tile_h & 0xFFFFu);            // tile_dim1 (tile_dim2=0)
    g1[5] = (int)(u32)stride_elems;                  // tensor_dim0_stride[31:0]
    g1[6] = 0;                                       // stride[47:32] | dim1_stride lo
    g1[7] = 0;
    v4i z4; z4[0] = z4[1] = z4[2] = z4[3] = 0;       // groups 2/3 unused (2D)
    v8i z8;
#pragma unroll
    for (int i = 0; i < 8; ++i) z8[i] = 0;
    __builtin_amdgcn_tensor_load_to_lds(g0, g1, z4, z4, z8, 0);
}
#endif

// dot product of two 96-element bf16 rows (4-byte packed reads)
__device__ __forceinline__ float dot96(const u16* qr, const u16* kr) {
    const u32* q2 = (const u32*)qr;
    const u32* k2 = (const u32*)kr;
    float acc = 0.f;
#pragma unroll
    for (int i = 0; i < 48; ++i) {
        u32 qa = q2[i], ka = k2[i];
        acc += bits2f(qa << 16) * bits2f(ka << 16);
        acc += bits2f(qa & 0xFFFF0000u) * bits2f(ka & 0xFFFF0000u);
    }
    return acc;
}

// ---------------------------------------------------------------------------
// Kernel 1: xb = bf16(x + props_pe)      (props broadcast over batch)
// ---------------------------------------------------------------------------
__global__ __launch_bounds__(256) void add_pe_bf16(const float* __restrict__ x,
                                                   const float* __restrict__ pe,
                                                   u16* __restrict__ xb,
                                                   int total, int SD) {
    int i4 = (blockIdx.x * 256 + threadIdx.x) * 4;
    if (i4 >= total) return;
    float4 a = *(const float4*)(x + i4);
    int p = i4 % SD;
    float4 b = *(const float4*)(pe + p);
    ushort4 o;
    o.x = f2bf(a.x + b.x); o.y = f2bf(a.y + b.y);
    o.z = f2bf(a.z + b.z); o.w = f2bf(a.w + b.w);
    *(ushort4*)(xb + i4) = o;
}

// Kernel 2: plain f32 -> bf16 convert (pos_prompt_emb)
__global__ __launch_bounds__(256) void conv_bf16(const float* __restrict__ src,
                                                 u16* __restrict__ dst, int total) {
    int i4 = (blockIdx.x * 256 + threadIdx.x) * 4;
    if (i4 >= total) return;
    float4 a = *(const float4*)(src + i4);
    ushort4 o;
    o.x = f2bf(a.x); o.y = f2bf(a.y); o.z = f2bf(a.z); o.w = f2bf(a.w);
    *(ushort4*)(dst + i4) = o;
}

// ---------------------------------------------------------------------------
// Kernel 3: WT[n][k] = bf16(W[k][n])   (768x768 tiled transpose + convert)
// ---------------------------------------------------------------------------
__global__ __launch_bounds__(256) void transpose_bf16(const float* __restrict__ W,
                                                      u16* __restrict__ WT, int dim) {
    __shared__ float tile[32][33];
    int tx = threadIdx.x, ty = threadIdx.y;       // 32 x 8
    int bx = blockIdx.x * 32, by = blockIdx.y * 32;
#pragma unroll
    for (int j = 0; j < 32; j += 8)
        tile[ty + j][tx] = W[(size_t)(by + ty + j) * dim + bx + tx];
    __syncthreads();
#pragma unroll
    for (int j = 0; j < 32; j += 8)
        WT[(size_t)(bx + ty + j) * dim + by + tx] = f2bf(tile[tx][ty + j]);
}

// ---------------------------------------------------------------------------
// Kernel 4: generic bf16 WMMA GEMM:  C[M,N] = A[M,K] * WT^T + bias
//   A    : bf16 row-major, stride K
//   WT   : bf16 [N][K]  (pre-transposed weights, contiguous along K)
//   out  : bf16 (outF32==0) or f32 (outF32==1), bias fused
// Block tile 128x128, 8 waves (2x4), wave tile 64x32, BK=32.
// Tile staging via Tensor Data Mover (TENSORcnt) when available; manual
// vector loads as fallback and for partial M tiles.
// ---------------------------------------------------------------------------
__global__ __launch_bounds__(256) void gemm_bf16_wmma(const u16* __restrict__ A,
                                                      const u16* __restrict__ WT,
                                                      const float* __restrict__ bias,
                                                      void* __restrict__ Cv,
                                                      int M, int N, int K, int outF32) {
    __shared__ __align__(16) u16 As[128 * 32];
    __shared__ __align__(16) u16 Bs[128 * 32];

    const int t    = threadIdx.x;
    const int lane = t & 31;
    const int w    = t >> 5;
    const int wm   = w >> 2;          // 0..1
    const int wn   = w & 3;           // 0..3
    const int bm   = blockIdx.x * 128;
    const int bn   = blockIdx.y * 128;
    const bool fullA = (bm + 128 <= M);

#if USE_TDM
    const u32 asAddr = lds_byte_addr(As);
    const u32 bsAddr = lds_byte_addr(Bs);
#endif

    v8f acc[4][2];
#pragma unroll
    for (int mi = 0; mi < 4; ++mi)
#pragma unroll
        for (int ni = 0; ni < 2; ++ni)
#pragma unroll
            for (int j = 0; j < 8; ++j) acc[mi][ni][j] = 0.f;

    for (int k0 = 0; k0 < K; k0 += 32) {
#if USE_TDM
        // ---- TDM tile staging: wave 0 issues 2D tile DMAs, waits TENSORcnt
        if (w == 0) {
            if (fullA)
                tdm_load_2d_bf16(asAddr, A + (size_t)bm * K + k0, K, 32, 128);
            tdm_load_2d_bf16(bsAddr, WT + (size_t)bn * K + k0, K, 32, 128);
            __builtin_amdgcn_s_wait_tensorcnt(0);
        }
        if (!fullA) {
            // distributed guarded A-tile load (zero-fill out-of-range rows)
#pragma unroll
            for (int rep = 0; rep < 2; ++rep) {
                int l   = t + rep * 256;
                int row = l >> 2;
                int c8  = (l & 3) * 8;
                int ga  = bm + row;
                uint4 av; av.x = av.y = av.z = av.w = 0u;
                if (ga < M) av = *(const uint4*)(A + (size_t)ga * K + k0 + c8);
                *(uint4*)(As + row * 32 + c8) = av;
            }
        }
#else
        // ---- manual cooperative loads: 128 rows x 32 cols bf16, 16B x2/thread
#pragma unroll
        for (int rep = 0; rep < 2; ++rep) {
            int l   = t + rep * 256;
            int row = l >> 2;
            int c8  = (l & 3) * 8;
            int ga  = bm + row;
            uint4 av; av.x = av.y = av.z = av.w = 0u;
            if (ga < M) av = *(const uint4*)(A + (size_t)ga * K + k0 + c8);
            *(uint4*)(As + row * 32 + c8) = av;
            uint4 bv = *(const uint4*)(WT + (size_t)(bn + row) * K + k0 + c8);
            *(uint4*)(Bs + row * 32 + c8) = bv;
            if (k0 + 32 < K) {                 // gfx1250 global_prefetch path
                if (ga < M) __builtin_prefetch(A + (size_t)ga * K + k0 + 32 + c8, 0, 1);
                __builtin_prefetch(WT + (size_t)(bn + row) * K + k0 + 32 + c8, 0, 1);
            }
        }
#endif
        __syncthreads();

        // ---- fragment loads (ISA 7.12.2 bf16 layouts)
        // A 16x32: lanes 0-15 -> K 0..7 & 16..23 ; lanes 16-31 -> K 8..15 & 24..31
        const int kA = (lane & 16) ? 8 : 0;
        // B 32x16: lanes 0-15 -> K 0..7 & 16..23 ; lanes 16-31 -> K 8..15 & 24..31 (col = lane&15)
        const int kB = (lane & 16) ? 8 : 0;

        FragBf16 af[4], bf[2];
#pragma unroll
        for (int mi = 0; mi < 4; ++mi) {
            int r = wm * 64 + mi * 16 + (lane & 15);
            af[mi].q[0] = *(const uint4*)(As + r * 32 + kA);
            af[mi].q[1] = *(const uint4*)(As + r * 32 + kA + 16);
        }
#pragma unroll
        for (int ni = 0; ni < 2; ++ni) {
            int r = wn * 32 + ni * 16 + (lane & 15);
            bf[ni].q[0] = *(const uint4*)(Bs + r * 32 + kB);
            bf[ni].q[1] = *(const uint4*)(Bs + r * 32 + kB + 16);
        }

#pragma unroll
        for (int mi = 0; mi < 4; ++mi)
#pragma unroll
            for (int ni = 0; ni < 2; ++ni)
                acc[mi][ni] = __builtin_amdgcn_wmma_f32_16x16x32_bf16(
                    false, af[mi].v, false, bf[ni].v,
                    (short)0, acc[mi][ni], false, false);
        __syncthreads();
    }

    // ---- epilogue: C layout lanes 0-15 -> M 0..7, lanes 16-31 -> M 8..15
    const int rowoff = (lane & 16) ? 8 : 0;
#pragma unroll
    for (int mi = 0; mi < 4; ++mi)
#pragma unroll
        for (int ni = 0; ni < 2; ++ni) {
            int col = bn + wn * 32 + ni * 16 + (lane & 15);
            float bv = bias[col];
#pragma unroll
            for (int j = 0; j < 8; ++j) {
                int row = bm + wm * 64 + mi * 16 + rowoff + j;
                if (row < M) {
                    float v = acc[mi][ni][j] + bv;
                    if (outF32) ((float*)Cv)[(size_t)row * N + col] = v;
                    else        ((u16*)Cv)[(size_t)row * N + col] = f2bf(v);
                }
            }
        }
}

// ---------------------------------------------------------------------------
// Kernel 5: attention, one workgroup per (b,h).
// Two passes over S in chunks of 64 with online softmax; computes pos_w,
// clamped & renormalized neg_w, and both weighted V sums. Outputs bf16 with
// head-interleaved layout pos_o[b, q, h*96+d].
// ---------------------------------------------------------------------------
__global__ __launch_bounds__(256) void attn_kernel(const u16* __restrict__ Qb,
                                                   const u16* __restrict__ Kb,
                                                   const u16* __restrict__ Vb,
                                                   u16* __restrict__ pos_o,
                                                   u16* __restrict__ neg_o) {
    const int b = blockIdx.x >> 3;
    const int h = blockIdx.x & 7;
    const int t = threadIdx.x;

    __shared__ __align__(16) u16 Qs[LQ * HD];     //  9216 B
    __shared__ float St[LQ * 64];                 // 12288 B
    __shared__ float NWt[LQ * 64];                // 12288 B
    __shared__ __align__(16) u16 Kt[64 * HD];     // 12288 B
    __shared__ __align__(16) u16 Vt[64 * HD];     // 12288 B
    __shared__ float mArr[LQ], lArr[LQ], nsum[LQ];

    // stage Q for this head (bf16): LQ rows of 96
    for (int i = t; i < LQ * HD / 2; i += 256) {
        int q = i / (HD / 2), kp = i % (HD / 2);
        ((u32*)Qs)[i] = *(const u32*)(Qb + q * DD + h * HD + kp * 2);
    }
    __syncthreads();

    // ------------------- pass 1: online max / sumexp --------------------
    float mm = -1e30f, ll = 0.f;
    for (int s0 = 0; s0 < SS; s0 += 64) {
        for (int i = t; i < 64 * 12; i += 256) {  // 64 rows x 12 uint4
            int r = i / 12, c = i % 12;
            *(uint4*)(Kt + r * HD + c * 8) =
                *(const uint4*)(Kb + (size_t)(b * SS + s0 + r) * DD + h * HD + c * 8);
        }
        __syncthreads();
        for (int i = t; i < LQ * 64; i += 256) {
            int q = i >> 6, s = i & 63;
            St[i] = QK_SCALE * dot96(Qs + q * HD, Kt + s * HD);
        }
        __syncthreads();
        if (t < LQ) {
            float cm = mm;
            for (int s = 0; s < 64; ++s) cm = fmaxf(cm, St[t * 64 + s]);
            float sum = 0.f;
            for (int s = 0; s < 64; ++s) sum += __expf(St[t * 64 + s] - cm);
            ll = ll * __expf(mm - cm) + sum;
            mm = cm;
        }
        __syncthreads();
    }
    if (t < LQ) { mArr[t] = mm; lArr[t] = ll; nsum[t] = 0.f; }
    __syncthreads();

    // ------------------- pass 2: weights + weighted sums ----------------
    float posAcc[18], negAcc[18];
#pragma unroll
    for (int i = 0; i < 18; ++i) { posAcc[i] = 0.f; negAcc[i] = 0.f; }
    const float invSm1 = 1.0f / (float)(SS - 1);

    for (int s0 = 0; s0 < SS; s0 += 64) {
        for (int i = t; i < 64 * 12; i += 256) {
            int r = i / 12, c = i % 12;
            size_t go = (size_t)(b * SS + s0 + r) * DD + h * HD + c * 8;
            *(uint4*)(Kt + r * HD + c * 8) = *(const uint4*)(Kb + go);
            *(uint4*)(Vt + r * HD + c * 8) = *(const uint4*)(Vb + go);
        }
        __syncthreads();
        for (int i = t; i < LQ * 64; i += 256) {
            int q = i >> 6, s = i & 63;
            float sc = QK_SCALE * dot96(Qs + q * HD, Kt + s * HD);
            float p  = __expf(sc - mArr[q]) / lArr[q];
            St[i] = p;
            float nw = fminf((1.f - p) * invSm1, 1e-6f / (p + 1e-10f));
            NWt[i] = nw;
            atomicAdd(&nsum[q], nw);
        }
        __syncthreads();
        // each thread owns 18 fixed (q,d) pairs: pair = i*256 + t
#pragma unroll
        for (int i = 0; i < 18; ++i) {
            int pair = i * 256 + t;
            int q = pair / HD, d = pair - q * HD;
            float pa = posAcc[i], na = negAcc[i];
            for (int s = 0; s < 64; ++s) {
                float v = bf2f(Vt[s * HD + d]);
                pa += St[q * 64 + s] * v;
                na += NWt[q * 64 + s] * v;
            }
            posAcc[i] = pa; negAcc[i] = na;
        }
        __syncthreads();
    }

#pragma unroll
    for (int i = 0; i < 18; ++i) {
        int pair = i * 256 + t;
        int q = pair / HD, d = pair - q * HD;
        size_t off = (size_t)(b * LQ + q) * DD + h * HD + d;
        pos_o[off] = f2bf(posAcc[i]);
        neg_o[off] = f2bf(negAcc[i] / nsum[q]);
    }
}

// ---------------------------------------------------------------------------
// Kernel 6: Householder orthogonalize + write outputs (pos first, then neg)
// ---------------------------------------------------------------------------
__global__ __launch_bounds__(256) void householder_kernel(const float* __restrict__ pos_p,
                                                          const float* __restrict__ neg_p,
                                                          float* __restrict__ out) {
    const int row = blockIdx.x;             // 0 .. B*LQ-1
    const int t = threadIdx.x;
    __shared__ float sxy[256], sxx[256];
    float xy = 0.f, xx = 0.f;
    for (int e = t; e < EE; e += 256) {
        float pp = pos_p[(size_t)row * EE + e];
        float nn = neg_p[(size_t)row * EE + e];
        xy += pp * nn; xx += pp * pp;
    }
    sxy[t] = xy; sxx[t] = xx;
    __syncthreads();
    for (int st = 128; st > 0; st >>= 1) {
        if (t < st) { sxy[t] += sxy[t + st]; sxx[t] += sxx[t + st]; }
        __syncthreads();
    }
    float lam = sxy[0] / (sxx[0] + 1e-8f);
    const size_t TOT = (size_t)BB * LQ * EE;
    for (int e = t; e < EE; e += 256) {
        float pp = pos_p[(size_t)row * EE + e];
        float nn = neg_p[(size_t)row * EE + e];
        out[(size_t)row * EE + e]       = pp;
        out[TOT + (size_t)row * EE + e] = nn - lam * pp;
    }
}

// ---------------------------------------------------------------------------
// host-side launcher
// ---------------------------------------------------------------------------
extern "C" void kernel_launch(void* const* d_in, const int* in_sizes, int n_in,
                              void* d_out, int out_size, void* d_ws, size_t ws_size,
                              hipStream_t stream) {
    (void)in_sizes; (void)n_in; (void)out_size; (void)ws_size;

    const float* x    = (const float*)d_in[0];
    const float* pe   = (const float*)d_in[1];
    const float* ppe  = (const float*)d_in[2];
    const float* Wq   = (const float*)d_in[3];  const float* bq = (const float*)d_in[4];
    const float* Wk   = (const float*)d_in[5];  const float* bk = (const float*)d_in[6];
    const float* Wv   = (const float*)d_in[7];  const float* bv = (const float*)d_in[8];
    const float* Wo   = (const float*)d_in[9];  const float* bo = (const float*)d_in[10];
    const float* Wp   = (const float*)d_in[11]; const float* bp = (const float*)d_in[12];
    float* out = (float*)d_out;

    char* ws = (char*)d_ws;
    constexpr size_t XB_BYTES = (size_t)BB * SS * DD * 2;      // 100,663,296
    constexpr size_t WT_BYTES = (size_t)DD * DD * 2;           //   1,179,648
    u16* xb  = (u16*)(ws);
    u16* Kb  = (u16*)(ws + XB_BYTES);
    u16* Vb  = (u16*)(ws + 2 * XB_BYTES);
    u16* WqT = (u16*)(ws + 3 * XB_BYTES);
    u16* WkT = (u16*)(ws + 3 * XB_BYTES + 1 * WT_BYTES);
    u16* WvT = (u16*)(ws + 3 * XB_BYTES + 2 * WT_BYTES);
    u16* WoT = (u16*)(ws + 3 * XB_BYTES + 3 * WT_BYTES);
    u16* WpT = (u16*)(ws + 3 * XB_BYTES + 4 * WT_BYTES);
    u16* ppb = (u16*)(ws + 3 * XB_BYTES + 5 * WT_BYTES);
    u16* Qb  = (u16*)(ws + 3 * XB_BYTES + 5 * WT_BYTES + 73728);
    // xb region is dead after the V projection -> alias attention/proj buffers
    u16*   pos_o  = (u16*)(ws);
    u16*   neg_o  = (u16*)(ws + WT_BYTES);
    u16*   pos_wo = (u16*)(ws + 2 * WT_BYTES);
    u16*   neg_wo = (u16*)(ws + 3 * WT_BYTES);
    float* pos_p  = (float*)(ws + 4 * WT_BYTES);
    float* neg_p  = (float*)(ws + 4 * WT_BYTES + (size_t)DD * DD * 4);

    const int total = BB * SS * DD;
    add_pe_bf16<<<(total / 4 + 255) / 256, 256, 0, stream>>>(x, pe, xb, total, SS * DD);
    conv_bf16<<<(LQ * DD / 4 + 255) / 256, 256, 0, stream>>>(ppe, ppb, LQ * DD);

    dim3 tb(32, 8), tg(DD / 32, DD / 32);
    transpose_bf16<<<tg, tb, 0, stream>>>(Wq, WqT, DD);
    transpose_bf16<<<tg, tb, 0, stream>>>(Wk, WkT, DD);
    transpose_bf16<<<tg, tb, 0, stream>>>(Wv, WvT, DD);
    transpose_bf16<<<tg, tb, 0, stream>>>(Wo, WoT, DD);
    transpose_bf16<<<tg, tb, 0, stream>>>(Wp, WpT, DD);

    // Q projection (M=48), K/V projections (M=65536)
    gemm_bf16_wmma<<<dim3(1, DD / 128), 256, 0, stream>>>(ppb, WqT, bq, Qb, LQ, DD, DD, 0);
    gemm_bf16_wmma<<<dim3(BB * SS / 128, DD / 128), 256, 0, stream>>>(xb, WkT, bk, Kb, BB * SS, DD, DD, 0);
    gemm_bf16_wmma<<<dim3(BB * SS / 128, DD / 128), 256, 0, stream>>>(xb, WvT, bv, Vb, BB * SS, DD, DD, 0);

    attn_kernel<<<BB * HH, 256, 0, stream>>>(Qb, Kb, Vb, pos_o, neg_o);

    // output projections: (pos|neg)_o @ Wo + bo, then @ Wp + bp
    dim3 g768(BB * LQ / 128, DD / 128);   // (6,6)
    gemm_bf16_wmma<<<g768, 256, 0, stream>>>(pos_o, WoT, bo, pos_wo, BB * LQ, DD, DD, 0);
    gemm_bf16_wmma<<<g768, 256, 0, stream>>>(neg_o, WoT, bo, neg_wo, BB * LQ, DD, DD, 0);
    gemm_bf16_wmma<<<g768, 256, 0, stream>>>(pos_wo, WpT, bp, pos_p, BB * LQ, EE, DD, 1);
    gemm_bf16_wmma<<<g768, 256, 0, stream>>>(neg_wo, WpT, bp, neg_p, BB * LQ, EE, DD, 1);

    householder_kernel<<<BB * LQ, 256, 0, stream>>>(pos_p, neg_p, out);
}
